// SimpleKeyboardGaAN_56564719288771
// MI455X (gfx1250) — matrix operand order repaired and checked by reading, not compiled, not run on hardware
//
#include <hip/hip_runtime.h>
#include <hip/hip_bf16.h>

// ---------------- problem constants ----------------
#define NN      50000        // nodes
#define NE      400000       // edges (before self loops)
#define ETOT    450000       // edges + self loops
#define INCH    814
#define K1PAD   832          // 26 * 32
#define HIDC    128
#define NH1     4            // heads, layer 1
#define HC1     512          // NH1 * HIDC
#define NG      64
#define NCLS    46
#define NEG_SLOPE 0.2f

typedef __attribute__((ext_vector_type(16))) __bf16 v16bf;
typedef __attribute__((ext_vector_type(8)))  float  v8f;
typedef __attribute__((ext_vector_type(4)))  int    i4x;   // trivial 16B vector

union Frag { v16bf v; i4x q[2]; };
union Pack16 { unsigned short s[16]; i4x q[2]; };

// f32 -> bf16 round-to-nearest-even
__device__ __forceinline__ unsigned short f2bf(float f) {
  unsigned u = __float_as_uint(f);
  u += 0x7FFFu + ((u >> 16) & 1u);
  return (unsigned short)(u >> 16);
}
// monotone order-preserving float<->u32 encoding for atomicMax
__device__ __forceinline__ unsigned encf(float f) {
  unsigned u = __float_as_uint(f);
  return (u & 0x80000000u) ? ~u : (u | 0x80000000u);
}
__device__ __forceinline__ float decf(unsigned u) {
  return __uint_as_float((u & 0x80000000u) ? (u & 0x7FFFFFFFu) : ~u);
}
__device__ __forceinline__ void get_edge(const int* __restrict__ ei, int e, int& s, int& d) {
  if (e < NE) { s = ei[e]; d = ei[NE + e]; } else { s = d = e - NE; }
}

// ---------------- W (KxN, f32) -> Wt (NxKpad, bf16, zero padded) ----------------
__global__ void transpose_w_kernel(const float* __restrict__ W, unsigned short* __restrict__ Wt,
                                   int K, int Nc, int Kpad) {
  long idx = (long)blockIdx.x * blockDim.x + threadIdx.x;
  if (idx >= (long)Nc * Kpad) return;
  int n = (int)(idx / Kpad), k = (int)(idx % Kpad);
  Wt[idx] = (k < K) ? f2bf(W[(long)k * Nc + n]) : (unsigned short)0;
}

// ---------------- bf16 WMMA GEMM: C[MxNcols] = A[MxK] * Bt^T  ----------------
// Bt is N-major bf16 [Ncols x Kpad].  AF32=1: A is f32 (converted on the fly).
template <int AF32>
__global__ __launch_bounds__(256) void gemm_bf16_kernel(const void* __restrict__ Aptr,
                                                        const unsigned short* __restrict__ Bt,
                                                        float* __restrict__ C,
                                                        int M, int K, int Kpad, int Ncols) {
  constexpr int STR = 40;                      // bf16 elems per LDS row (80B, conflict-free)
  __shared__ unsigned short As[128 * STR];     // 128 x 32 tile
  __shared__ unsigned short Bs[128 * STR];     // 128(N) x 32(K) tile, N-major

  const int tid  = threadIdx.x;
  const int lane = tid & 31, wid = tid >> 5;
  const int wm = (wid & 3) * 32, wn = (wid >> 2) * 64;   // wave sub-tile origin
  const int fm = lane & 15, fg = lane >> 4;              // WMMA lane decomposition
  const int srow = tid >> 1, scol = (tid & 1) * 16;      // staging: 2 threads/row
  const long arow = (long)blockIdx.x * 128 + srow;
  const long brow = (long)blockIdx.y * 128 + srow;
  const bool rowok = (arow < M);

  const v8f zero = {};
  const i4x zero4 = {};
  v8f acc[2][4];
#pragma unroll
  for (int i = 0; i < 2; ++i)
#pragma unroll
    for (int j = 0; j < 4; ++j) acc[i][j] = zero;

  const int ksteps = Kpad / 32;
  for (int kt = 0; kt < ksteps; ++kt) {
    const int k0 = kt * 32;
    __syncthreads();
    // ---- stage A tile (convert f32->bf16 if needed); no per-element branches ----
    Pack16 ta;
    ta.q[0] = zero4; ta.q[1] = zero4;
    if (AF32) {
      const float* A = (const float*)Aptr;
      const long base = arow * (long)K;
      if (rowok) {                      // single divergent region
        if (k0 + 32 <= K) {             // wave-uniform: fully in-bounds (25/26 steps)
          const float2* ap = (const float2*)(A + base + k0 + scol);  // 8B aligned
#pragma unroll
          for (int j = 0; j < 8; ++j) {
            float2 t = ap[j];
            ta.s[2 * j]     = f2bf(t.x);
            ta.s[2 * j + 1] = f2bf(t.y);
          }
          __builtin_prefetch(A + base + k0 + 32 + scol, 0, 1);  // speculative, drop-safe
        } else {                        // K tail: branchless clamp + select
#pragma unroll
          for (int j = 0; j < 16; ++j) {
            int k = k0 + scol + j;
            float v = A[base + (k < K ? k : 0)];
            ta.s[j] = f2bf(k < K ? v : 0.f);
          }
        }
      }
    } else {
      const unsigned short* A = (const unsigned short*)Aptr;
      if (rowok) {
        const i4x* ap = (const i4x*)(A + arow * (long)Kpad + k0 + scol);
        ta.q[0] = ap[0]; ta.q[1] = ap[1];
      }
    }
    *(i4x*)&As[srow * STR + scol]     = ta.q[0];
    *(i4x*)&As[srow * STR + scol + 8] = ta.q[1];
    // ---- stage B tile (already bf16, N-major, zero-padded to Kpad) ----
    {
      const i4x* bp = (const i4x*)(Bt + brow * (long)Kpad + k0 + scol);
      *(i4x*)&Bs[srow * STR + scol]     = bp[0];
      *(i4x*)&Bs[srow * STR + scol + 8] = bp[1];
      if (kt + 1 < ksteps) __builtin_prefetch(bp + 4, 0, 1);
    }
    __syncthreads();
    // ---- fragments from LDS (all ds_load_b128, 16B aligned) ----
    Frag a[2], b[4];
#pragma unroll
    for (int mt = 0; mt < 2; ++mt) {
      int base = (wm + mt * 16 + fm) * STR;
      a[mt].q[0] = *(const i4x*)&As[base + fg * 8];       // K = fg*8 .. +7
      a[mt].q[1] = *(const i4x*)&As[base + 16 + fg * 8];  // K = 16+fg*8 .. +7
    }
#pragma unroll
    for (int nt = 0; nt < 4; ++nt) {
      int base = (wn + nt * 16 + fm) * STR + fg * 8;      // K = fg*8 .. +7, 16+fg*8 .. +7
      b[nt].q[0] = *(const i4x*)&Bs[base];
      b[nt].q[1] = *(const i4x*)&Bs[base + 16];
    }
#pragma unroll
    for (int mt = 0; mt < 2; ++mt)
#pragma unroll
      for (int nt = 0; nt < 4; ++nt)
        acc[mt][nt] = __builtin_amdgcn_wmma_f32_16x16x32_bf16(
            false, a[mt].v, false, b[nt].v, (short)0, acc[mt][nt], false, false);
  }
  // ---- store C per 16x16 f32 C/D layout: VGPR r -> (M = 8*fg + r, N = fm) ----
#pragma unroll
  for (int mt = 0; mt < 2; ++mt)
#pragma unroll
    for (int nt = 0; nt < 4; ++nt)
#pragma unroll
      for (int r = 0; r < 8; ++r) {
        long grow = (long)blockIdx.x * 128 + wm + mt * 16 + fg * 8 + r;
        int  gcol = blockIdx.y * 128 + wn + nt * 16 + fm;
        if (grow < M) C[grow * (long)Ncols + gcol] = acc[mt][nt][r];
      }
}

// ---------------- attention alpha_src / alpha_dst: wave per (node, head) ----------------
__global__ void attn_al_kernel(const float* __restrict__ Hm, const float* __restrict__ a_s,
                               const float* __restrict__ a_d, float* __restrict__ als,
                               float* __restrict__ ald, int N, int Hh, int HC) {
  int node = blockIdx.x * 8 + (threadIdx.x >> 5);
  if (node >= N) return;
  int h = blockIdx.y, lane = threadIdx.x & 31;
  const float* hp  = Hm  + (long)node * HC + h * HIDC;
  const float* asp = a_s + h * HIDC;
  const float* adp = a_d + h * HIDC;
  float ss = 0.f, sd = 0.f;
#pragma unroll
  for (int j = 0; j < 4; ++j) {
    float v = hp[lane + 32 * j];
    ss += v * asp[lane + 32 * j];
    sd += v * adp[lane + 32 * j];
  }
#pragma unroll
  for (int off = 16; off; off >>= 1) { ss += __shfl_xor(ss, off); sd += __shfl_xor(sd, off); }
  if (lane == 0) { als[node * Hh + h] = ss; ald[node * Hh + h] = sd; }
}

// ---------------- per-edge score + segment max (encoded u32 atomicMax) ----------------
__global__ void edge_score_kernel(const int* __restrict__ ei, const float* __restrict__ als,
                                  const float* __restrict__ ald, float* __restrict__ esc,
                                  unsigned* __restrict__ mx, int Hh) {
  int e = blockIdx.x * blockDim.x + threadIdx.x;
  if (e >= ETOT) return;
  int h = blockIdx.y, s, d;
  get_edge(ei, e, s, d);
  float v = als[s * Hh + h] + ald[d * Hh + h];
  float lr = v > 0.f ? v : NEG_SLOPE * v;
  esc[(long)e * Hh + h] = lr;
  atomicMax(&mx[d * Hh + h], encf(lr));
}

// ---------------- exp(e - max) + segment sum ----------------
__global__ void edge_exp_kernel(const int* __restrict__ ei, float* __restrict__ esc,
                                const unsigned* __restrict__ mx, float* __restrict__ den, int Hh) {
  int e = blockIdx.x * blockDim.x + threadIdx.x;
  if (e >= ETOT) return;
  int h = blockIdx.y, s, d;
  get_edge(ei, e, s, d);
  float ex = __expf(esc[(long)e * Hh + h] - decf(mx[d * Hh + h]));
  esc[(long)e * Hh + h] = ex;
  atomicAdd(&den[d * Hh + h], ex);
}

// ---------------- aggregation: wave per (edge, head), out[dst] += H[src]*alpha ----------------
__global__ void edge_agg_kernel(const int* __restrict__ ei, const float* __restrict__ esc,
                                const float* __restrict__ den, const float* __restrict__ Hm,
                                float* __restrict__ out, int Hh, int HC) {
  int e = blockIdx.x * 8 + (threadIdx.x >> 5);
  if (e >= ETOT) return;
  int h = blockIdx.y, lane = threadIdx.x & 31, s, d;
  get_edge(ei, e, s, d);
  float alpha = esc[(long)e * Hh + h] / den[d * Hh + h];
  const float* hp = Hm + (long)s * HC + h * HIDC;
  float* op = out + (long)d * HC + h * HIDC;
#pragma unroll
  for (int j = 0; j < 4; ++j) atomicAdd(&op[lane + 32 * j], hp[lane + 32 * j] * alpha);
}

// ---------------- bias + ReLU + bf16 convert (layer-2 GEMM input) ----------------
__global__ void bias_relu_bf16_kernel(const float* __restrict__ in, const float* __restrict__ bias,
                                      unsigned short* __restrict__ outb, long total, int colmask) {
  long idx = (long)blockIdx.x * blockDim.x + threadIdx.x;
  if (idx >= total) return;
  int c = (int)(idx & colmask);
  outb[idx] = f2bf(fmaxf(in[idx] + bias[c], 0.f));
}

// ---------------- pooling: relu(out2 + b2) scatter-added per graph ----------------
__global__ void pool_kernel(const float* __restrict__ out2, const float* __restrict__ b2,
                            const int* __restrict__ batch, float* __restrict__ pooled) {
  long idx = (long)blockIdx.x * blockDim.x + threadIdx.x;
  if (idx >= (long)NN * HIDC) return;
  int n = (int)(idx >> 7), c = (int)(idx & 127);
  float v = fmaxf(out2[idx] + b2[c], 0.f);
  atomicAdd(&pooled[batch[n] * HIDC + c], v);
}
__global__ void count_kernel(const int* __restrict__ batch, float* __restrict__ cnt) {
  int n = blockIdx.x * blockDim.x + threadIdx.x;
  if (n < NN) atomicAdd(&cnt[batch[n]], 1.f);
}

// ---------------- tiny MLP head ----------------
__global__ void fc1_kernel(const float* __restrict__ pooled, const float* __restrict__ cnt,
                           const float* __restrict__ w, const float* __restrict__ b,
                           float* __restrict__ z) {
  int idx = blockIdx.x * blockDim.x + threadIdx.x;
  if (idx >= NG * 64) return;
  int g = idx >> 6, j = idx & 63;
  float inv = 1.f / fmaxf(cnt[g], 1.f);
  float acc = b[j];
  for (int c = 0; c < HIDC; ++c) acc += pooled[g * HIDC + c] * inv * w[c * 64 + j];
  z[idx] = fmaxf(acc, 0.f);
}
__global__ void fc2_kernel(const float* __restrict__ z, const float* __restrict__ w,
                           const float* __restrict__ b, float* __restrict__ out) {
  int idx = blockIdx.x * blockDim.x + threadIdx.x;
  if (idx >= NG * NCLS) return;
  int g = idx / NCLS, k = idx % NCLS;
  float acc = b[k];
  for (int j = 0; j < 64; ++j) acc += z[g * 64 + j] * w[j * NCLS + k];
  out[idx] = acc;
}

// ---------------- host orchestration ----------------
extern "C" void kernel_launch(void* const* d_in, const int* in_sizes, int n_in,
                              void* d_out, int out_size, void* d_ws, size_t ws_size,
                              hipStream_t stream) {
  const float* x      = (const float*)d_in[0];
  const int*   ei     = (const int*)d_in[1];
  const int*   batch  = (const int*)d_in[2];
  const float* W1     = (const float*)d_in[3];
  const float* a_src1 = (const float*)d_in[4];
  const float* a_dst1 = (const float*)d_in[5];
  const float* b1     = (const float*)d_in[6];
  const float* W2     = (const float*)d_in[7];
  const float* a_src2 = (const float*)d_in[8];
  const float* a_dst2 = (const float*)d_in[9];
  const float* b2     = (const float*)d_in[10];
  const float* fc1_w  = (const float*)d_in[11];
  const float* fc1_b  = (const float*)d_in[12];
  const float* fc2_w  = (const float*)d_in[13];
  const float* fc2_b  = (const float*)d_in[14];
  float* out = (float*)d_out;

  // ---- workspace carve-out (256B aligned) ----
  char* p = (char*)d_ws;
  auto alloc = [&](size_t bytes) { void* r = (void*)p; p += (bytes + 255) & ~(size_t)255; return r; };
  unsigned short* w1t  = (unsigned short*)alloc((size_t)HC1 * K1PAD * 2);   // 512 x 832 bf16
  unsigned short* w2t  = (unsigned short*)alloc((size_t)HIDC * HC1 * 2);    // 128 x 512 bf16
  float* Hbuf1 = (float*)alloc((size_t)NN * HC1 * 4);                       // H1, reused as H2
  float* obuf1 = (float*)alloc((size_t)NN * HC1 * 4);                       // out1, reused as out2
  unsigned short* h1b = (unsigned short*)alloc((size_t)NN * HC1 * 2);
  float*    als1 = (float*)alloc((size_t)NN * NH1 * 4);
  float*    ald1 = (float*)alloc((size_t)NN * NH1 * 4);
  unsigned* mx1  = (unsigned*)alloc((size_t)NN * NH1 * 4);
  float*    den1 = (float*)alloc((size_t)NN * NH1 * 4);
  float*    esc1 = (float*)alloc((size_t)ETOT * NH1 * 4);
  float*    als2 = (float*)alloc((size_t)NN * 4);
  float*    ald2 = (float*)alloc((size_t)NN * 4);
  unsigned* mx2  = (unsigned*)alloc((size_t)NN * 4);
  float*    den2 = (float*)alloc((size_t)NN * 4);
  float*    esc2 = (float*)alloc((size_t)ETOT * 4);
  float* pooled = (float*)alloc((size_t)NG * HIDC * 4);
  float* cnt    = (float*)alloc((size_t)NG * 4);
  float* z      = (float*)alloc((size_t)NG * 64 * 4);
  float* Hbuf2 = Hbuf1;   // alias: H1 dead after layer-1 epilogue
  float* obuf2 = obuf1;   // alias: out1 dead after layer-1 epilogue

  // ---- weight transpose+convert ----
  transpose_w_kernel<<<((size_t)HC1 * K1PAD + 255) / 256, 256, 0, stream>>>(W1, w1t, INCH, HC1, K1PAD);
  transpose_w_kernel<<<((size_t)HIDC * HC1 + 255) / 256, 256, 0, stream>>>(W2, w2t, HC1, HIDC, HC1);

  // ---- layer 1 ----
  gemm_bf16_kernel<1><<<dim3((NN + 127) / 128, HC1 / 128), 256, 0, stream>>>(
      (const void*)x, w1t, Hbuf1, NN, INCH, K1PAD, HC1);
  attn_al_kernel<<<dim3((NN + 7) / 8, NH1), 256, 0, stream>>>(Hbuf1, a_src1, a_dst1, als1, ald1, NN, NH1, HC1);
  (void)hipMemsetAsync(mx1, 0, (size_t)NN * NH1 * 4, stream);   // 0 < enc(-inf) <= any score encoding
  (void)hipMemsetAsync(den1, 0, (size_t)NN * NH1 * 4, stream);
  (void)hipMemsetAsync(obuf1, 0, (size_t)NN * HC1 * 4, stream);
  edge_score_kernel<<<dim3((ETOT + 255) / 256, NH1), 256, 0, stream>>>(ei, als1, ald1, esc1, mx1, NH1);
  edge_exp_kernel<<<dim3((ETOT + 255) / 256, NH1), 256, 0, stream>>>(ei, esc1, mx1, den1, NH1);
  edge_agg_kernel<<<dim3((ETOT + 7) / 8, NH1), 256, 0, stream>>>(ei, esc1, den1, Hbuf1, obuf1, NH1, HC1);
  bias_relu_bf16_kernel<<<((size_t)NN * HC1 + 255) / 256, 256, 0, stream>>>(
      obuf1, b1, h1b, (long)NN * HC1, HC1 - 1);

  // ---- layer 2 (Hbuf2/obuf2 alias layer-1 buffers) ----
  gemm_bf16_kernel<0><<<dim3((NN + 127) / 128, HIDC / 128), 256, 0, stream>>>(
      (const void*)h1b, w2t, Hbuf2, NN, HC1, HC1, HIDC);
  attn_al_kernel<<<dim3((NN + 7) / 8, 1), 256, 0, stream>>>(Hbuf2, a_src2, a_dst2, als2, ald2, NN, 1, HIDC);
  (void)hipMemsetAsync(mx2, 0, (size_t)NN * 4, stream);
  (void)hipMemsetAsync(den2, 0, (size_t)NN * 4, stream);
  (void)hipMemsetAsync(obuf2, 0, (size_t)NN * HIDC * 4, stream);
  edge_score_kernel<<<dim3((ETOT + 255) / 256, 1), 256, 0, stream>>>(ei, als2, ald2, esc2, mx2, 1);
  edge_exp_kernel<<<dim3((ETOT + 255) / 256, 1), 256, 0, stream>>>(ei, esc2, mx2, den2, 1);
  edge_agg_kernel<<<dim3((ETOT + 7) / 8, 1), 256, 0, stream>>>(ei, esc2, den2, Hbuf2, obuf2, 1, HIDC);

  // ---- pooling + MLP head ----
  (void)hipMemsetAsync(pooled, 0, (size_t)NG * HIDC * 4, stream);
  (void)hipMemsetAsync(cnt, 0, (size_t)NG * 4, stream);
  pool_kernel<<<((size_t)NN * HIDC + 255) / 256, 256, 0, stream>>>(obuf2, b2, batch, pooled);
  count_kernel<<<(NN + 255) / 256, 256, 0, stream>>>(batch, cnt);
  fc1_kernel<<<(NG * 64 + 255) / 256, 256, 0, stream>>>(pooled, cnt, fc1_w, fc1_b, z);
  fc2_kernel<<<(NG * NCLS + 255) / 256, 256, 0, stream>>>(z, fc2_w, fc2_b, out);

  (void)in_sizes; (void)n_in; (void)out_size; (void)ws_size;
}
